// CustomMultiHeadAttention_34789235098342
// MI455X (gfx1250) — matrix operand order repaired and checked by reading, not compiled, not run on hardware
//
#include <hip/hip_runtime.h>
#include <hip/hip_bf16.h>

// ---------------------------------------------------------------------------
// AutoCorrelation attention for MI455X (gfx1250, wave32, WMMA bf16).
//   B=16, L=2048, D=512, H=8, DK=64, TOP_K=7
// Pipeline:
//   1) convert weights fp32->bf16
//   2) q_s/k_s/v_s = X @ W^T + b           (WMMA bf16 GEMM, bf16 out)
//   3) mean_value[b,tau] += diag-reduced q_s[b] @ k_s[b]^T  (WMMA + LDS bins)
//   4) top-7 of batch-mean, per-batch softmax (with 1/512 scale)
//   5) context = sum_i w_i * roll(v_s, -idx_i)
//   6) out = context @ fc_w^T + fc_b        (WMMA bf16 GEMM, fp32 out)
// ---------------------------------------------------------------------------

typedef __bf16  v16bf __attribute__((ext_vector_type(16)));
typedef __bf16  v8bf  __attribute__((ext_vector_type(8)));
typedef float   v8f   __attribute__((ext_vector_type(8)));

#define DMODEL 512
#define SEQL   2048
#define NBATCH 16
#define TOPK   7
#define TILE_M 128
#define TILE_N 64
#define KSTEP  32
#define LDSR   40   // padded LDS row stride in halves (80B, 16B-aligned, conflict-free)

__device__ __forceinline__ float bf2f(unsigned short h) {
    unsigned int u = ((unsigned int)h) << 16;
    float f; __builtin_memcpy(&f, &u, 4); return f;
}
__device__ __forceinline__ unsigned short f2bf(float f) {
    unsigned int u; __builtin_memcpy(&u, &f, 4);
    u += 0x7fffu + ((u >> 16) & 1u);            // round-to-nearest-even
    return (unsigned short)(u >> 16);
}

__device__ __forceinline__ v8f wmma_bf16(v16bf a, v16bf b, v8f c) {
    return __builtin_amdgcn_wmma_f32_16x16x32_bf16(
        /*neg_a=*/false, a, /*neg_b=*/false, b,
        /*c_mod=*/(short)0, c, /*reuse_a=*/false, /*reuse_b=*/false);
}

// ---- LDS tile loaders (256 threads). X tile: 128 x 32; W tile: 64 x 32 ----
__device__ __forceinline__ void load_x_f32(const float* __restrict__ X,
                                           int m0, int k0, unsigned short* ldsX) {
    int t = threadIdx.x;
    int row = t >> 1, cb = (t & 1) * 16;
    const float* src = X + (size_t)(m0 + row) * DMODEL + k0 + cb;
    if (k0 + KSTEP < DMODEL) __builtin_prefetch(src + KSTEP, 0, 0); // global_prefetch_b8
    unsigned short* dst = ldsX + row * LDSR + cb;
#pragma unroll
    for (int i = 0; i < 4; ++i) {
        float4 f = ((const float4*)src)[i];
        dst[4*i+0] = f2bf(f.x); dst[4*i+1] = f2bf(f.y);
        dst[4*i+2] = f2bf(f.z); dst[4*i+3] = f2bf(f.w);
    }
}
__device__ __forceinline__ void load_x_bf16(const unsigned short* __restrict__ X,
                                            int m0, int k0, unsigned short* ldsX) {
    int t = threadIdx.x;
    int row = t >> 1, cb = (t & 1) * 16;
    const unsigned short* src = X + (size_t)(m0 + row) * DMODEL + k0 + cb;
    if (k0 + KSTEP < DMODEL) __builtin_prefetch(src + KSTEP, 0, 0);
    unsigned short* dst = ldsX + row * LDSR + cb;
    *(uint4*)dst       = *(const uint4*)src;
    *(uint4*)(dst + 8) = *(const uint4*)(src + 8);
}
__device__ __forceinline__ void load_w_bf16(const unsigned short* __restrict__ W,
                                            int n0, int k0, unsigned short* ldsW) {
    int t = threadIdx.x;
    int row = t >> 2, cb = (t & 3) * 8;
    const unsigned short* src = W + (size_t)(n0 + row) * DMODEL + k0 + cb;
    if (k0 + KSTEP < DMODEL) __builtin_prefetch(src + KSTEP, 0, 0);
    *(uint4*)(ldsW + row * LDSR + cb) = *(const uint4*)src;
}

// ---- per-wave 32x32 MACs from LDS tiles (CDNA5 bf16 fragment layouts) ----
__device__ __forceinline__ void frag_mma(const unsigned short* ldsX,
                                         const unsigned short* ldsW,
                                         int wm, int wn, int lane, v8f acc[4]) {
    int l16   = lane & 15;
    int hi8   = (lane >> 4) & 1;
    int koffA = hi8 * 8;    // A: halves [0..7]=K{0..7|8..15}, [8..15]=K{16..23|24..31}
    int koffB = hi8 * 8;    // B: same K order as A

    const unsigned short* xr0 = ldsX + (wm * 32 + l16) * LDSR;
    const unsigned short* xr1 = xr0 + 16 * LDSR;
    const unsigned short* wr0 = ldsW + (wn * 32 + l16) * LDSR;
    const unsigned short* wr1 = wr0 + 16 * LDSR;

    v16bf a0, a1, b0, b1;
    ((v8bf*)&a0)[0] = *(const v8bf*)(xr0 + koffA);
    ((v8bf*)&a0)[1] = *(const v8bf*)(xr0 + 16 + koffA);
    ((v8bf*)&a1)[0] = *(const v8bf*)(xr1 + koffA);
    ((v8bf*)&a1)[1] = *(const v8bf*)(xr1 + 16 + koffA);
    ((v8bf*)&b0)[0] = *(const v8bf*)(wr0 + koffB);
    ((v8bf*)&b0)[1] = *(const v8bf*)(wr0 + 16 + koffB);
    ((v8bf*)&b1)[0] = *(const v8bf*)(wr1 + koffB);
    ((v8bf*)&b1)[1] = *(const v8bf*)(wr1 + 16 + koffB);

    acc[0] = wmma_bf16(a0, b0, acc[0]);
    acc[1] = wmma_bf16(a0, b1, acc[1]);
    acc[2] = wmma_bf16(a1, b0, acc[2]);
    acc[3] = wmma_bf16(a1, b1, acc[3]);
}

// ---- generic Y = X @ W^T + bias ------------------------------------------
template<bool XF32, bool OUTF32>
__global__ __launch_bounds__(256) void gemm_xwT_kernel(const void* __restrict__ Xv,
                                                       const unsigned short* __restrict__ W,
                                                       const float* __restrict__ bias,
                                                       void* __restrict__ Yv) {
    __shared__ __attribute__((aligned(16))) unsigned short ldsX[TILE_M * LDSR];
    __shared__ __attribute__((aligned(16))) unsigned short ldsW[TILE_N * LDSR];
    int m0 = blockIdx.x * TILE_M;
    int n0 = blockIdx.y * TILE_N;
    int lane = threadIdx.x & 31, wv = threadIdx.x >> 5;
    int wm = wv >> 1, wn = wv & 1;

    v8f zero = {0.f,0.f,0.f,0.f,0.f,0.f,0.f,0.f};
    v8f acc[4] = {zero, zero, zero, zero};

    for (int k0 = 0; k0 < DMODEL; k0 += KSTEP) {
        __syncthreads();
        if constexpr (XF32) load_x_f32((const float*)Xv, m0, k0, ldsX);
        else                load_x_bf16((const unsigned short*)Xv, m0, k0, ldsX);
        load_w_bf16(W, n0, k0, ldsW);
        __syncthreads();
        frag_mma(ldsX, ldsW, wm, wn, lane, acc);
    }

    int hi8 = (lane >> 4) & 1, l16 = lane & 15;
    int gm = m0 + wm * 32 + hi8 * 8;
    int gn = n0 + wn * 32 + l16;
#pragma unroll
    for (int ai = 0; ai < 4; ++ai) {
        int mf = (ai >> 1) * 16, nf = (ai & 1) * 16;
        int n = gn + nf;
        float bv = bias[n];
#pragma unroll
        for (int r = 0; r < 8; ++r) {
            int m = gm + mf + r;
            float y = acc[ai][r] + bv;
            if constexpr (OUTF32) ((float*)Yv)[(size_t)m * DMODEL + n] = y;
            else ((unsigned short*)Yv)[(size_t)m * DMODEL + n] = f2bf(y);
        }
    }
}

// ---- G = q_s[b] @ k_s[b]^T folded into circular diagonals -----------------
__global__ __launch_bounds__(256) void corr_kernel(const unsigned short* __restrict__ q_s,
                                                   const unsigned short* __restrict__ k_s,
                                                   float* __restrict__ mv) {
    __shared__ __attribute__((aligned(16))) unsigned short ldsX[TILE_M * LDSR];
    __shared__ __attribute__((aligned(16))) unsigned short ldsW[TILE_N * LDSR];
    __shared__ float bins[8][64];

    int b = blockIdx.z;
    const unsigned short* X = q_s + (size_t)b * SEQL * DMODEL;
    const unsigned short* W = k_s + (size_t)b * SEQL * DMODEL;
    int m0 = blockIdx.x * TILE_M;   // i tile
    int n0 = blockIdx.y * TILE_N;   // j tile
    int lane = threadIdx.x & 31, wv = threadIdx.x >> 5;
    int wm = wv >> 1, wn = wv & 1;

    v8f zero = {0.f,0.f,0.f,0.f,0.f,0.f,0.f,0.f};
    v8f acc[4] = {zero, zero, zero, zero};

    for (int k0 = 0; k0 < DMODEL; k0 += KSTEP) {
        __syncthreads();
        load_x_bf16(X, m0, k0, ldsX);
        load_w_bf16(W, n0, k0, ldsW);
        __syncthreads();
        frag_mma(ldsX, ldsW, wm, wn, lane, acc);
    }

    // per-wave 63 diagonal bins in LDS (wave-private -> no cross-wave sync)
    bins[wv][lane] = 0.f; bins[wv][lane + 32] = 0.f;
    int hi8 = (lane >> 4) & 1, l16 = lane & 15;
    int dbase = (m0 + wm * 32) - (n0 + wn * 32);   // i0 - j0 of this wave tile
#pragma unroll
    for (int ai = 0; ai < 4; ++ai) {
        int mf = (ai >> 1) * 16, nf = (ai & 1) * 16;
#pragma unroll
        for (int r = 0; r < 8; ++r) {
            int dl = (mf - nf) + (hi8 * 8 + r) - l16 + 31;   // 0..62
            atomicAdd(&bins[wv][dl], acc[ai][r]);            // ds_add_f32
        }
    }
    float* mvb = mv + (size_t)b * SEQL;
    for (int t = lane; t < 63; t += 32) {
        float v = bins[wv][t];
        if (v != 0.f)
            atomicAdd(&mvb[(dbase + t - 31) & (SEQL - 1)], v);
    }
}

// ---- batch-mean + iterative top-7 ----------------------------------------
__global__ void topk_kernel(const float* __restrict__ mv, float* __restrict__ meanB,
                            int* __restrict__ idx_out) {
    __shared__ float sv[256];
    __shared__ int   si[256];
    int tid = threadIdx.x;
    for (int t = tid; t < SEQL; t += 256) {
        float s = 0.f;
        for (int b = 0; b < NBATCH; ++b) s += mv[b * SEQL + t];
        meanB[t] = s;
    }
    __syncthreads();
    for (int it = 0; it < TOPK; ++it) {
        float best = -3.4e38f; int bi = 0;
        for (int t = tid; t < SEQL; t += 256) {
            float v = meanB[t];
            if (v > best) { best = v; bi = t; }
        }
        sv[tid] = best; si[tid] = bi;
        __syncthreads();
        for (int s = 128; s > 0; s >>= 1) {
            if (tid < s && sv[tid + s] > sv[tid]) { sv[tid] = sv[tid + s]; si[tid] = si[tid + s]; }
            __syncthreads();
        }
        if (tid == 0) { idx_out[it] = si[0]; meanB[si[0]] = -3.4e38f; }
        __syncthreads();
    }
}

// ---- per-batch softmax over the 7 selected lags (scale 1/512 first!) ------
__global__ void softmax_kernel(const float* __restrict__ mv, const int* __restrict__ idx,
                               float* __restrict__ wts) {
    int b = threadIdx.x;
    if (b >= NBATCH) return;
    float v[TOPK]; float mx = -3.4e38f;
#pragma unroll
    for (int i = 0; i < TOPK; ++i) {
        v[i] = mv[b * SEQL + idx[i]] * (1.f / (float)DMODEL);
        mx = fmaxf(mx, v[i]);
    }
    float s = 0.f;
#pragma unroll
    for (int i = 0; i < TOPK; ++i) { v[i] = __expf(v[i] - mx); s += v[i]; }
    float inv = 1.f / s;
#pragma unroll
    for (int i = 0; i < TOPK; ++i) wts[b * TOPK + i] = v[i] * inv;
}

// ---- context[b,t,:] = sum_i w[b,i] * v_s[b,(t+idx_i)%L,:] -----------------
__global__ __launch_bounds__(128) void context_kernel(const unsigned short* __restrict__ v_s,
                                                      const int* __restrict__ idx,
                                                      const float* __restrict__ wts,
                                                      unsigned short* __restrict__ ctx) {
    int bt = blockIdx.x;
    int b = bt >> 11;            // / SEQL
    int t = bt & (SEQL - 1);
    int c = threadIdx.x * 4;
    float a0 = 0.f, a1 = 0.f, a2 = 0.f, a3 = 0.f;
#pragma unroll
    for (int i = 0; i < TOPK; ++i) {
        int src = (t + idx[i]) & (SEQL - 1);
        const unsigned short* vp = v_s + (((size_t)b * SEQL + src) * DMODEL + c);
        uint2 u = *(const uint2*)vp;
        float w = wts[b * TOPK + i];
        a0 += w * bf2f((unsigned short)(u.x & 0xffffu));
        a1 += w * bf2f((unsigned short)(u.x >> 16));
        a2 += w * bf2f((unsigned short)(u.y & 0xffffu));
        a3 += w * bf2f((unsigned short)(u.y >> 16));
    }
    uint2 o;
    o.x = (unsigned)f2bf(a0) | ((unsigned)f2bf(a1) << 16);
    o.y = (unsigned)f2bf(a2) | ((unsigned)f2bf(a3) << 16);
    *(uint2*)(ctx + (((size_t)b * SEQL + t) * DMODEL + c)) = o;
}

// ---- fp32 -> bf16 weight conversion --------------------------------------
__global__ void f32_to_bf16_kernel(const float* __restrict__ src,
                                   unsigned short* __restrict__ dst, int n) {
    int i = blockIdx.x * blockDim.x + threadIdx.x;
    if (i < n) dst[i] = f2bf(src[i]);
}

// ---------------------------------------------------------------------------
extern "C" void kernel_launch(void* const* d_in, const int* in_sizes, int n_in,
                              void* d_out, int out_size, void* d_ws, size_t ws_size,
                              hipStream_t stream) {
    (void)in_sizes; (void)n_in; (void)out_size; (void)ws_size;
    const float* Q    = (const float*)d_in[0];
    const float* K    = (const float*)d_in[1];
    const float* V    = (const float*)d_in[2];
    const float* WQ_w = (const float*)d_in[3];
    const float* WQ_b = (const float*)d_in[4];
    const float* WK_w = (const float*)d_in[5];
    const float* WK_b = (const float*)d_in[6];
    const float* WV_w = (const float*)d_in[7];
    const float* WV_b = (const float*)d_in[8];
    const float* FC_w = (const float*)d_in[9];
    const float* FC_b = (const float*)d_in[10];

    char* ws = (char*)d_ws;
    const size_t SZ_W   = (size_t)DMODEL * DMODEL * 2;          // 512 KB
    const size_t SZ_ACT = (size_t)NBATCH * SEQL * DMODEL * 2;   // 32 MB
    unsigned short* wqb  = (unsigned short*)(ws);
    unsigned short* wkb  = (unsigned short*)(ws + SZ_W);
    unsigned short* wvb  = (unsigned short*)(ws + 2 * SZ_W);
    unsigned short* fcb  = (unsigned short*)(ws + 3 * SZ_W);
    unsigned short* q_s  = (unsigned short*)(ws + 4 * SZ_W);
    unsigned short* k_s  = (unsigned short*)(ws + 4 * SZ_W + SZ_ACT);
    unsigned short* v_s  = (unsigned short*)(ws + 4 * SZ_W + 2 * SZ_ACT);
    float*          mv   = (float*)        (ws + 4 * SZ_W + 3 * SZ_ACT);
    float*          mB   = (float*)        (ws + 4 * SZ_W + 3 * SZ_ACT + NBATCH * SEQL * 4);
    int*            idx  = (int*)          (ws + 4 * SZ_W + 3 * SZ_ACT + NBATCH * SEQL * 4 + SEQL * 4);
    float*          wts  = (float*)        (ws + 4 * SZ_W + 3 * SZ_ACT + NBATCH * SEQL * 4 + SEQL * 4 + 256);
    unsigned short* ctx  = q_s;   // q_s dead after corr_kernel -> reuse for context

    const int NW = DMODEL * DMODEL;
    f32_to_bf16_kernel<<<NW / 256, 256, 0, stream>>>(WQ_w, wqb, NW);
    f32_to_bf16_kernel<<<NW / 256, 256, 0, stream>>>(WK_w, wkb, NW);
    f32_to_bf16_kernel<<<NW / 256, 256, 0, stream>>>(WV_w, wvb, NW);
    f32_to_bf16_kernel<<<NW / 256, 256, 0, stream>>>(FC_w, fcb, NW);

    dim3 gproj(NBATCH * SEQL / TILE_M, DMODEL / TILE_N);   // (256, 8)
    gemm_xwT_kernel<true, false><<<gproj, 256, 0, stream>>>(Q, wqb, WQ_b, q_s);
    gemm_xwT_kernel<true, false><<<gproj, 256, 0, stream>>>(K, wkb, WK_b, k_s);
    gemm_xwT_kernel<true, false><<<gproj, 256, 0, stream>>>(V, wvb, WV_b, v_s);

    hipMemsetAsync(mv, 0, (size_t)NBATCH * SEQL * 4, stream);
    dim3 gcorr(SEQL / TILE_M, SEQL / TILE_N, NBATCH);      // (16, 32, 16)
    corr_kernel<<<gcorr, 256, 0, stream>>>(q_s, k_s, mv);

    topk_kernel<<<1, 256, 0, stream>>>(mv, mB, idx);
    softmax_kernel<<<1, 32, 0, stream>>>(mv, idx, wts);

    context_kernel<<<NBATCH * SEQL, 128, 0, stream>>>(v_s, idx, wts, ctx);

    gemm_xwT_kernel<false, true><<<gproj, 256, 0, stream>>>(ctx, fcb, FC_b, (float*)d_out);
}
